// Encoder_51092930953761
// MI455X (gfx1250) — matrix operand order repaired and checked by reading, not compiled, not run on hardware
//
#include <hip/hip_runtime.h>
#include <hip/hip_bf16.h>

// ---------------------------------------------------------------------------
// Transformer encoder layer for MI455X (gfx1250), bf16 WMMA everywhere.
// B=4, S=2048, EMB=1024, HEADS=16, HEAD_DIM=64, FF=4096
// ---------------------------------------------------------------------------

#define EMBD 1024
#define NHEAD 16
#define HDIM 64
#define BATCH 4
#define SEQ 2048
#define FFD 4096
#define MROWS (BATCH * SEQ) // 8192

typedef __bf16 bf16_t;
typedef bf16_t v16bf __attribute__((ext_vector_type(16)));
typedef bf16_t v8bf __attribute__((ext_vector_type(8)));
typedef float v8f __attribute__((ext_vector_type(8)));

union BF16x16 { v16bf v; v8bf h[2]; };
union BF8pack { v8bf v; int i[4]; };

#define WMMA_BF16(A, B, C) \
  __builtin_amdgcn_wmma_f32_16x16x32_bf16(false, (A), false, (B), (short)0, (C), false, false)

__device__ __forceinline__ bf16_t f2bf(float f) {
  union { float f; unsigned u; } a; a.f = f;
  unsigned short s = (unsigned short)((a.u + 0x7fffu + ((a.u >> 16) & 1u)) >> 16);
  union { unsigned short s; bf16_t b; } o; o.s = s;
  return o.b;
}

// SWAPX16 across the two 16-lane halves of a wave32 (group-of-32 xor swizzle).
__device__ __forceinline__ float swz16(float x) {
  return __int_as_float(__builtin_amdgcn_ds_swizzle(__float_as_int(x), 0x401f));
}

// CDNA5 async copy: global -> LDS, 16B per lane, tracked by ASYNCcnt.
// lds_off is the 32-bit LDS byte offset (low 32 bits of the generic address).
__device__ __forceinline__ void async_ld_b128(unsigned lds_off, const void* gptr) {
  asm volatile("global_load_async_to_lds_b128 %0, %1, off"
               :: "v"(lds_off), "v"(gptr)
               : "memory");
}
__device__ __forceinline__ void wait_async0() {
  asm volatile("s_wait_asynccnt 0" ::: "memory");
}

// ---------------------------------------------------------------------------
// fp32 -> bf16 conversion
// ---------------------------------------------------------------------------
__global__ void cvt_bf16(const float* __restrict__ src, bf16_t* __restrict__ dst, int n) {
  int i = blockIdx.x * 256 + threadIdx.x;
  if (i < n) dst[i] = f2bf(src[i]);
}

// ---------------------------------------------------------------------------
// Generic C[M,N] = A[M,K](bf16) @ W[N,K](bf16)^T + bias, WMMA bf16.
// Block: 256 threads (8 waves). Per wave: 32 rows x 64 cols (8 WMMA tiles
// per 32-wide k-step).  The 64x32 B tile is shared by all 8 waves and is
// staged through LDS with double-buffered GLOBAL_LOAD_ASYNC_TO_LDS_B128
// (one 16B chunk per thread per k-step).  Last k-step is peeled so the
// steady-state loop is branchless.
// mode 0: outB = bf16((acc+bias)*scale)
// mode 1: outF = acc+bias
// mode 2: outB = bf16(gelu(acc+bias))   (exact gelu, erf-based)
// ---------------------------------------------------------------------------
__global__ void __launch_bounds__(256) gemm_bt(
    const bf16_t* __restrict__ A, const bf16_t* __restrict__ W,
    const float* __restrict__ bias, float* __restrict__ outF,
    bf16_t* __restrict__ outB, int M, int N, int K, int mode, float scale) {
  __shared__ __align__(16) unsigned char smem[2 * 4096];  // double-buffered B tile

  const int tid = threadIdx.x;
  const int wave = tid >> 5;
  const int lane = tid & 31;
  const int hi = lane >> 4;
  const int li = lane & 15;
  const int m0 = blockIdx.y * 256 + wave * 32;
  const int n0 = blockIdx.x * 64;

  // B staging: thread -> (row 0..63, 16B part 0..3) of the 64x32 bf16 tile
  const int srow = tid >> 2;
  const int spart = tid & 3;
  const bf16_t* gB = W + (size_t)(n0 + srow) * K + spart * 8;
  const unsigned ldsBase = (unsigned)(size_t)&smem[0];
  const unsigned ldsWr = ldsBase + (unsigned)(srow * 64 + spart * 16);

  // A operands: two 16-row tiles per wave
  const bf16_t* Ar0 = A + (size_t)(m0 + li) * K + 16 * hi;
  const bf16_t* Ar1 = Ar0 + (size_t)16 * K;

  v8f acc[2][4] = {};

  auto compute_step = [&](int k0, int cur) __attribute__((always_inline)) {
    BF16x16 a0, a1;
    a0.h[0] = *(const v8bf*)(Ar0 + k0);       // k = 16*hi .. 16*hi+7
    a0.h[1] = *(const v8bf*)(Ar0 + k0 + 8);   // k = 16*hi+8 .. 16*hi+15
    a1.h[0] = *(const v8bf*)(Ar1 + k0);
    a1.h[1] = *(const v8bf*)(Ar1 + k0 + 8);
    const bf16_t* sB = (const bf16_t*)(smem + cur * 4096);
#pragma unroll
    for (int t = 0; t < 4; t++) {
      v16bf bm = *(const v16bf*)(sB + (16 * t + li) * 32 + 16 * hi);
      acc[0][t] = WMMA_BF16(a0.v, bm, acc[0][t]);
      acc[1][t] = WMMA_BF16(a1.v, bm, acc[1][t]);
    }
  };

  // prologue: stage k0 = 0 into buffer 0
  async_ld_b128(ldsWr, gB);

  int k0 = 0;
  for (; k0 < K - 32; k0 += 32) {  // steady state: always prefetch next
    const int cur = (k0 >> 5) & 1;
    wait_async0();
    __syncthreads();  // B tile for this k-step is visible to all waves
    async_ld_b128(ldsWr + (unsigned)((1 - cur) * 4096), gB + k0 + 32);
    compute_step(k0, cur);
    __syncthreads();  // all waves done reading before this buffer is re-filled
  }
  {  // peeled last k-step: no prefetch, no trailing barrier
    const int cur = (k0 >> 5) & 1;
    wait_async0();
    __syncthreads();
    compute_step(k0, cur);
  }

  float biasv[4];
#pragma unroll
  for (int t = 0; t < 4; t++) biasv[t] = bias[n0 + 16 * t + li];

#pragma unroll
  for (int mt = 0; mt < 2; mt++) {
#pragma unroll
    for (int t = 0; t < 4; t++) {
#pragma unroll
      for (int r = 0; r < 8; r++) {
        size_t idx = (size_t)(m0 + 16 * mt + 8 * hi + r) * N + (n0 + 16 * t + li);
        float val = acc[mt][t][r] + biasv[t];
        if (mode == 1) {
          outF[idx] = val;
        } else if (mode == 0) {
          outB[idx] = f2bf(val * scale);
        } else {
          float gv = 0.5f * val * (1.0f + erff(val * 0.70710678118654752f));
          outB[idx] = f2bf(gv);
        }
      }
    }
  }
}

// ---------------------------------------------------------------------------
// V [B,S,EMB] (bf16, head-major cols) -> Vt [B,H,HDIM,S] (bf16)
// ---------------------------------------------------------------------------
__global__ void transpose_v(const bf16_t* __restrict__ v, bf16_t* __restrict__ vt) {
  size_t i = (size_t)blockIdx.x * 256 + threadIdx.x;  // index into vt
  int s = (int)(i % SEQ);
  size_t t = i / SEQ;
  int d = (int)(t % HDIM);
  size_t t2 = t / HDIM;
  int h = (int)(t2 % NHEAD);
  int b = (int)(t2 / NHEAD);
  vt[i] = v[(size_t)(b * SEQ + s) * EMBD + h * HDIM + d];
}

// ---------------------------------------------------------------------------
// Flash attention, one wave per 16-query tile.
// scores^T = Kblk @ q^T  ->  each lane owns one query column; softmax
// reductions are 8-elt local + one SWAPX16 swizzle.  out^T = V^T @ P^T.
// q is pre-scaled by 0.125*log2(e), so exp2 is the softmax exponential.
// grid: (SEQ/128, NHEAD, BATCH), block: 256 (8 waves).
// ---------------------------------------------------------------------------
__global__ void __launch_bounds__(256) flash_attn(
    const bf16_t* __restrict__ q, const bf16_t* __restrict__ k,
    const bf16_t* __restrict__ vt, const int* __restrict__ mask,
    bf16_t* __restrict__ ctx) {
  const int wave = threadIdx.x >> 5;
  const int lane = threadIdx.x & 31;
  const int hi = lane >> 4;
  const int li = lane & 15;  // query col for B/C; row index for A operands
  const int b = blockIdx.z, h = blockIdx.y;
  const int q0 = (blockIdx.x * 8 + wave) * 16;

  // q as B-operand (column li = query q0+li), d split 0..31 / 32..63
  const bf16_t* qrow = q + (size_t)(b * SEQ + q0 + li) * EMBD + h * HDIM + 16 * hi;
  v16bf qb0 = *(const v16bf*)(qrow);
  v16bf qb1 = *(const v16bf*)(qrow + 32);

  v8f acc[4] = {};
  float mrun = -1e30f, lsum = 0.f;
  const int* mrow = mask + b * SEQ;
  const bf16_t* vbase = vt + (size_t)((b * NHEAD + h) * HDIM + li) * SEQ + 16 * hi;

  for (int j0 = 0; j0 < SEQ; j0 += 32) {
    // K rows as A-operand: tile0 = keys j0..j0+15, tile1 = keys j0+16..j0+31
    const bf16_t* kr0 = k + (size_t)(b * SEQ + j0 + li) * EMBD + h * HDIM + 16 * hi;
    const bf16_t* kr1 = kr0 + (size_t)16 * EMBD;
    BF16x16 ka0d0, ka0d1, ka1d0, ka1d1;
    ka0d0.h[0] = *(const v8bf*)(kr0);
    ka0d0.h[1] = *(const v8bf*)(kr0 + 8);
    ka0d1.h[0] = *(const v8bf*)(kr0 + 32);
    ka0d1.h[1] = *(const v8bf*)(kr0 + 40);
    ka1d0.h[0] = *(const v8bf*)(kr1);
    ka1d0.h[1] = *(const v8bf*)(kr1 + 8);
    ka1d1.h[0] = *(const v8bf*)(kr1 + 32);
    ka1d1.h[1] = *(const v8bf*)(kr1 + 40);

    v8f s0 = {}, s1 = {};
    s0 = WMMA_BF16(ka0d0.v, qb0, s0);
    s0 = WMMA_BF16(ka0d1.v, qb1, s0);
    s1 = WMMA_BF16(ka1d0.v, qb0, s1);
    s1 = WMMA_BF16(ka1d1.v, qb1, s1);

    // mask + running max (lane covers keys {8hi+r} and {16+8hi+r})
    float sv0[8], sv1[8];
    float bmax = -1e30f;
#pragma unroll
    for (int r = 0; r < 8; r++) {
      float a0 = mrow[j0 + 8 * hi + r] ? s0[r] : -1e30f;
      float a1 = mrow[j0 + 16 + 8 * hi + r] ? s1[r] : -1e30f;
      sv0[r] = a0;
      sv1[r] = a1;
      bmax = fmaxf(bmax, fmaxf(a0, a1));
    }
    bmax = fmaxf(bmax, swz16(bmax));
    float mnew = fmaxf(mrun, bmax);
    float sc = exp2f(mrun - mnew);

    float ls = 0.f;
    BF8pack p0, p1;
#pragma unroll
    for (int r = 0; r < 8; r++) {
      float e0 = exp2f(sv0[r] - mnew);
      float e1 = exp2f(sv1[r] - mnew);
      ls += e0 + e1;
      p0.v[r] = f2bf(e0);
      p1.v[r] = f2bf(e1);
    }
    mrun = mnew;
    lsum = lsum * sc + ls + swz16(ls);
#pragma unroll
    for (int t = 0; t < 4; t++) acc[t] *= sc;

    // Re-pack P into B-operand layout (K=32 keys x 16 queries):
    // lo lanes need keys 0..15 (own p0 ++ partner p0),
    // hi lanes need keys 16..31 (partner p1 ++ own p1).
    BF8pack p0x, p1x;
#pragma unroll
    for (int i2 = 0; i2 < 4; i2++) {
      p0x.i[i2] = __builtin_amdgcn_ds_swizzle(p0.i[i2], 0x401f);
      p1x.i[i2] = __builtin_amdgcn_ds_swizzle(p1.i[i2], 0x401f);
    }
    BF16x16 pb;
    pb.h[0] = hi ? p1x.v : p0.v;
    pb.h[1] = hi ? p1.v : p0x.v;

    // out^T += V^T tile (A-operand, contiguous from Vt) @ P^T
    const bf16_t* vr = vbase + j0;
#pragma unroll
    for (int t = 0; t < 4; t++) {
      BF16x16 va;
      va.h[0] = *(const v8bf*)(vr + (size_t)(16 * t) * SEQ);
      va.h[1] = *(const v8bf*)(vr + (size_t)(16 * t) * SEQ + 8);
      acc[t] = WMMA_BF16(va.v, pb.v, acc[t]);
    }
  }

  // ctx[query, h*64 + d], out^T tile t: d = 16t + 8hi + r, col = query = li
  float inv = 1.0f / lsum;
  bf16_t* crow = ctx + (size_t)(b * SEQ + q0 + li) * EMBD + h * HDIM + 8 * hi;
#pragma unroll
  for (int t = 0; t < 4; t++) {
    BF8pack o;
#pragma unroll
    for (int r = 0; r < 8; r++) o.v[r] = f2bf(acc[t][r] * inv);
    *(v8bf*)(crow + 16 * t) = o.v;
  }
}

// ---------------------------------------------------------------------------
// out = LayerNorm(x + y) * g + be ; optional bf16 copy. One block per row.
// ---------------------------------------------------------------------------
__global__ void __launch_bounds__(256) resid_ln(
    const float* __restrict__ x, const float* __restrict__ y,
    const float* __restrict__ g, const float* __restrict__ be,
    float* __restrict__ outF, bf16_t* __restrict__ outB) {
  __shared__ float red[256];
  const int tid = threadIdx.x;
  const size_t base = (size_t)blockIdx.x * EMBD;

  float v[4];
  float s = 0.f;
#pragma unroll
  for (int i = 0; i < 4; i++) {
    int c = tid + 256 * i;
    v[i] = x[base + c] + y[base + c];
    s += v[i];
  }
  red[tid] = s;
  __syncthreads();
  for (int off = 128; off > 0; off >>= 1) {
    if (tid < off) red[tid] += red[tid + off];
    __syncthreads();
  }
  float mu = red[0] * (1.0f / EMBD);
  __syncthreads();

  float s2 = 0.f;
#pragma unroll
  for (int i = 0; i < 4; i++) {
    float d = v[i] - mu;
    s2 += d * d;
  }
  red[tid] = s2;
  __syncthreads();
  for (int off = 128; off > 0; off >>= 1) {
    if (tid < off) red[tid] += red[tid + off];
    __syncthreads();
  }
  float rs = rsqrtf(red[0] * (1.0f / EMBD) + 1e-5f);

#pragma unroll
  for (int i = 0; i < 4; i++) {
    int c = tid + 256 * i;
    float o = (v[i] - mu) * rs * g[c] + be[c];
    outF[base + c] = o;
    if (outB) outB[base + c] = f2bf(o);
  }
}

// ---------------------------------------------------------------------------
// Host side
// ---------------------------------------------------------------------------
extern "C" void kernel_launch(void* const* d_in, const int* in_sizes, int n_in,
                              void* d_out, int out_size, void* d_ws, size_t ws_size,
                              hipStream_t stream) {
  (void)in_sizes; (void)n_in; (void)out_size; (void)ws_size;
  const float* x = (const float*)d_in[0];
  const int* mask = (const int*)d_in[1];
  const float* Wq = (const float*)d_in[2];
  const float* bq = (const float*)d_in[3];
  const float* Wk = (const float*)d_in[4];
  const float* bk = (const float*)d_in[5];
  const float* Wv = (const float*)d_in[6];
  const float* bv = (const float*)d_in[7];
  const float* Wo = (const float*)d_in[8];
  const float* bo = (const float*)d_in[9];
  const float* W1 = (const float*)d_in[10];
  const float* b1 = (const float*)d_in[11];
  const float* W2 = (const float*)d_in[12];
  const float* b2 = (const float*)d_in[13];
  const float* g1 = (const float*)d_in[14];
  const float* be1 = (const float*)d_in[15];
  const float* g2 = (const float*)d_in[16];
  const float* be2 = (const float*)d_in[17];

  // -------- workspace layout (lifetime-based reuse, ~168 MB total) --------
  char* ws = (char*)d_ws;
  const size_t MB = (size_t)1 << 20;
  bf16_t* Wq_b = (bf16_t*)(ws + 0 * MB);    // 2 MB
  bf16_t* Wk_b = (bf16_t*)(ws + 2 * MB);    // 2 MB
  bf16_t* Wv_b = (bf16_t*)(ws + 4 * MB);    // 2 MB
  bf16_t* Wo_b = (bf16_t*)(ws + 6 * MB);    // 2 MB
  bf16_t* W1_b = (bf16_t*)(ws + 8 * MB);    // 8 MB
  bf16_t* W2_b = (bf16_t*)(ws + 16 * MB);   // 8 MB
  bf16_t* Xb   = (bf16_t*)(ws + 24 * MB);   // 16 MB: Xbf -> ctx -> h_bf16
  bf16_t* ctx  = Xb;
  bf16_t* hbf  = Xb;
  bf16_t* qbuf = (bf16_t*)(ws + 40 * MB);   // 16 MB (region reused as ff1, 64 MB)
  bf16_t* kbuf = (bf16_t*)(ws + 56 * MB);   // 16 MB (part of ff1 region)
  bf16_t* vbuf = (bf16_t*)(ws + 72 * MB);   // 16 MB (part of ff1 region)
  bf16_t* ff1  = (bf16_t*)(ws + 40 * MB);   // 64 MB, after q/k/v are dead
  bf16_t* Vt   = (bf16_t*)(ws + 104 * MB);  // 16 MB -> attnout (32 MB) -> ff2 (32 MB)
  float* attnout = (float*)(ws + 104 * MB);
  float* ff2     = (float*)(ws + 104 * MB);
  float* hf      = (float*)(ws + 136 * MB); // 32 MB
  float* out     = (float*)d_out;

  const int nX = MROWS * EMBD;  // 8388608

  // 1) bf16 conversions
  cvt_bf16<<<nX / 256, 256, 0, stream>>>(x, Xb, nX);
  cvt_bf16<<<(EMBD * EMBD) / 256, 256, 0, stream>>>(Wq, Wq_b, EMBD * EMBD);
  cvt_bf16<<<(EMBD * EMBD) / 256, 256, 0, stream>>>(Wk, Wk_b, EMBD * EMBD);
  cvt_bf16<<<(EMBD * EMBD) / 256, 256, 0, stream>>>(Wv, Wv_b, EMBD * EMBD);
  cvt_bf16<<<(EMBD * EMBD) / 256, 256, 0, stream>>>(Wo, Wo_b, EMBD * EMBD);
  cvt_bf16<<<(FFD * EMBD) / 256, 256, 0, stream>>>(W1, W1_b, FFD * EMBD);
  cvt_bf16<<<(EMBD * FFD) / 256, 256, 0, stream>>>(W2, W2_b, EMBD * FFD);

  // 2) QKV projections (q pre-scaled by 1/sqrt(64) * log2(e) for exp2 softmax)
  dim3 blk(256);
  dim3 gE(EMBD / 64, MROWS / 256);  // (16, 32)
  const float qscale = 0.125f * 1.44269504088896f;
  gemm_bt<<<gE, blk, 0, stream>>>(Xb, Wq_b, bq, nullptr, qbuf, MROWS, EMBD, EMBD, 0, qscale);
  gemm_bt<<<gE, blk, 0, stream>>>(Xb, Wk_b, bk, nullptr, kbuf, MROWS, EMBD, EMBD, 0, 1.0f);
  gemm_bt<<<gE, blk, 0, stream>>>(Xb, Wv_b, bv, nullptr, vbuf, MROWS, EMBD, EMBD, 0, 1.0f);

  // 3) V -> Vt, flash attention -> ctx (reuses Xb region)
  transpose_v<<<nX / 256, 256, 0, stream>>>(vbuf, Vt);
  flash_attn<<<dim3(SEQ / 128, NHEAD, BATCH), blk, 0, stream>>>(qbuf, kbuf, Vt, mask, ctx);

  // 4) output projection (f32), residual + LN1 -> hf (f32) + hbf (bf16)
  gemm_bt<<<gE, blk, 0, stream>>>(ctx, Wo_b, bo, attnout, nullptr, MROWS, EMBD, EMBD, 1, 1.0f);
  resid_ln<<<MROWS, 256, 0, stream>>>(x, attnout, g1, be1, hf, hbf);

  // 5) FFN: gelu(h @ W1^T + b1) @ W2^T + b2
  gemm_bt<<<dim3(FFD / 64, MROWS / 256), blk, 0, stream>>>(hbf, W1_b, b1, nullptr, ff1,
                                                           MROWS, FFD, EMBD, 2, 1.0f);
  gemm_bt<<<dim3(EMBD / 64, MROWS / 256), blk, 0, stream>>>(ff1, W2_b, b2, ff2, nullptr,
                                                            MROWS, EMBD, FFD, 1, 1.0f);

  // 6) residual + LN2 -> d_out
  resid_ln<<<MROWS, 256, 0, stream>>>(hf, ff2, g2, be2, out, nullptr);
}
